// LlamaAttention_69441031241907
// MI455X (gfx1250) — matrix operand order repaired and checked by reading, not compiled, not run on hardware
//
#include <hip/hip_runtime.h>
#include <hip/hip_bf16.h>
#include <type_traits>

typedef __attribute__((ext_vector_type(16))) _Float16 v16h;
typedef __attribute__((ext_vector_type(8)))  _Float16 v8h;
typedef __attribute__((ext_vector_type(4)))  _Float16 v4h;
typedef __attribute__((ext_vector_type(8)))  float    v8f;
typedef __attribute__((ext_vector_type(4)))  int      v4i;

constexpr int B_    = 2;
constexpr int S_    = 2048;
constexpr int H_    = 4096;
constexpr int NH_   = 32;
constexpr int NKV_  = 8;
constexpr int HD_   = 128;
constexpr int QS_   = NH_ * HD_;            // 4096
constexpr int KVS_  = NKV_ * HD_;           // 1024
constexpr int QKVN_ = QS_ + 2 * KVS_;       // 6144
constexpr int MTOT  = B_ * S_;              // 4096

#if __has_builtin(__builtin_amdgcn_global_load_async_to_lds_b128)
#define HAVE_ASYNC_LDS 1
#endif

__device__ __forceinline__ void wait_ds0() {
#if __has_builtin(__builtin_amdgcn_s_wait_dscnt)
  __builtin_amdgcn_s_wait_dscnt(0);
#else
  asm volatile("s_wait_dscnt 0" ::: "memory");
#endif
}

__device__ __forceinline__ void wait_async0() {
#ifdef HAVE_ASYNC_LDS
#if __has_builtin(__builtin_amdgcn_s_wait_asynccnt)
  __builtin_amdgcn_s_wait_asynccnt(0);
#else
  asm volatile("s_wait_asynccnt 0" ::: "memory");
#endif
#endif
}

// 16-byte global -> LDS copy: async (ASYNCcnt-tracked) when available.
__device__ __forceinline__ void cp_b128(const _Float16* g, _Float16* l) {
#ifdef HAVE_ASYNC_LDS
  __builtin_amdgcn_global_load_async_to_lds_b128(
      (__attribute__((address_space(1))) v4i*)(void*)g,
      (__attribute__((address_space(3))) v4i*)(void*)l,
      0, 0);
#else
  *(v8h*)l = *(const v8h*)g;
#endif
}

// Fragment-ready k-swizzle: chunks reordered [0-7, 16-23, 8-15, 24-31] so a
// lane's 16 operand halves (K = 8*half + e for e < 8, 8*half + 8 + e for e >= 8, the WMMA A/B table order) are one
// contiguous, 32B-aligned LDS read.
__device__ __forceinline__ int kswz(int k) {
  return (k & 7) | ((k & 8) << 1) | ((k & 16) >> 1);
}

// ---------------------------------------------------------------------------
// fp32 -> fp16 conversion
// ---------------------------------------------------------------------------
__global__ __launch_bounds__(256)
void cvt_f32_f16(const float* __restrict__ src, _Float16* __restrict__ dst, size_t n) {
  size_t i = ((size_t)blockIdx.x * blockDim.x + threadIdx.x) * 4;
  if (i >= n) return;
  float4 v = *(const float4*)(src + i);
  v4h o;
  o[0] = (_Float16)v.x; o[1] = (_Float16)v.y;
  o[2] = (_Float16)v.z; o[3] = (_Float16)v.w;
  *(v4h*)(dst + i) = o;
}

// ---------------------------------------------------------------------------
// f16 WMMA GEMM: C[M,N] = A[M,K] x B[K,N], row-major. 256 threads = 8 waves,
// block tile 128x128, K-tile 32, double-buffered LDS, async A staging.
// ---------------------------------------------------------------------------
template <int OUT_F16>
__global__ __launch_bounds__(256)
void gemm_wmma(const _Float16* __restrict__ A, const _Float16* __restrict__ Bm,
               void* __restrict__ Cout, int M, int N, int K) {
  __shared__ __align__(32) _Float16 As[2][128 * 32];   // [row][kswz]
  __shared__ __align__(32) _Float16 Bs[2][128 * 32];   // [n][kswz]

  const int tid  = threadIdx.x;
  const int wave = tid >> 5;
  const int lane = tid & 31;
  const int half = lane >> 4;
  const int l16  = lane & 15;
  const int m0 = blockIdx.y * 128;
  const int n0 = blockIdx.x * 128;

  v8f acc[8] = {};

  auto stage = [&](int buf, int k0) {
    {
      // A tile: 128 rows x 32 halves, 2 x b128 async copies per thread.
      const int row = tid >> 1, hi = tid & 1;
      const _Float16* g = A + (size_t)(m0 + row) * K + k0 + hi * 16;
      _Float16* l = &As[buf][row * 32 + hi * 8];
      cp_b128(g, l);           // k0-7   or k16-23
      cp_b128(g + 8, l + 16);  // k8-15  or k24-31
    }
    {
      // B tile: 32 k-rows x 128 cols, transposed + k-swizzled into LDS.
      const int kk = tid >> 3, nb = (tid & 7) * 16;
      const int kp = kswz(kk);
      const v8h* g = (const v8h*)(Bm + (size_t)(k0 + kk) * N + n0 + nb);
      v8h b0 = g[0], b1 = g[1];
#pragma unroll
      for (int e = 0; e < 8; ++e) {
        Bs[buf][(nb + e) * 32 + kp]     = b0[e];
        Bs[buf][(nb + 8 + e) * 32 + kp] = b1[e];
      }
    }
  };

  stage(0, 0);
  wait_async0();
  __syncthreads();

  const int nk = K >> 5;
  for (int ki = 0; ki < nk; ++ki) {
    const int buf = ki & 1;
    const int k0  = ki << 5;
    if (ki + 1 < nk) stage(buf ^ 1, k0 + 32);
    if (ki + 2 < nk) {
      __builtin_prefetch(A + (size_t)(m0 + (tid >> 1)) * K + k0 + 64, 0, 1);
      __builtin_prefetch(Bm + (size_t)(k0 + 64 + (tid >> 3)) * N + n0 + (tid & 7) * 16, 0, 1);
    }

    const v16h af = *(const v16h*)(&As[buf][(wave * 16 + l16) * 32 + half * 16]);
#pragma unroll
    for (int t = 0; t < 8; ++t) {
      const v16h bf = *(const v16h*)(&Bs[buf][(t * 16 + l16) * 32 + half * 16]);
      acc[t] = __builtin_amdgcn_wmma_f32_16x16x32_f16(
          false, af, false, bf, (short)0, acc[t], false, false);
    }
    wait_async0();
    __syncthreads();
  }

  if (OUT_F16) {
    _Float16* C = (_Float16*)Cout;
#pragma unroll
    for (int t = 0; t < 8; ++t)
#pragma unroll
      for (int j = 0; j < 8; ++j)
        C[(size_t)(m0 + wave * 16 + j + half * 8) * N + n0 + t * 16 + l16] =
            (_Float16)acc[t][j];
  } else {
    float* C = (float*)Cout;
#pragma unroll
    for (int t = 0; t < 8; ++t)
#pragma unroll
      for (int j = 0; j < 8; ++j)
        C[(size_t)(m0 + wave * 16 + j + half * 8) * N + n0 + t * 16 + l16] =
            acc[t][j];
  }
}

// ---------------------------------------------------------------------------
// RoPE + head split. qkv16: [B*S, 6144] f16 -> q16 [B,NH,S,HD], k16 [B,NKV,S,HD]
// ---------------------------------------------------------------------------
__global__ __launch_bounds__(256)
void rope_split(const int* __restrict__ pos, const _Float16* __restrict__ qkv,
                _Float16* __restrict__ qo, _Float16* __restrict__ ko) {
  size_t i = (size_t)blockIdx.x * blockDim.x + threadIdx.x;
  const size_t total = (size_t)B_ * S_ * (NH_ + NKV_) * 64;
  if (i >= total) return;
  const int d = (int)(i & 63);
  size_t t = i >> 6;
  const int hh = (int)(t % (NH_ + NKV_)); t /= (NH_ + NKV_);
  const int s = (int)(t % S_);
  const int b = (int)(t / S_);

  const float p  = (float)pos[(size_t)b * S_ + s];
  const float fr = p * __expf(-(float)d * 0.14391156831212787f);  // 10000^(-d/64)
  float sn, cs;
  __sincosf(fr, &sn, &cs);

  const _Float16* src = qkv + ((size_t)b * S_ + s) * QKVN_;
  if (hh < NH_) {
    const _Float16* q = src + hh * HD_;
    const float x1 = (float)q[d], x2 = (float)q[d + 64];
    _Float16* dst = qo + (((size_t)b * NH_ + hh) * S_ + s) * HD_;
    dst[d]      = (_Float16)(x1 * cs - x2 * sn);
    dst[d + 64] = (_Float16)(x2 * cs + x1 * sn);
  } else {
    const int kh = hh - NH_;
    const _Float16* k = src + QS_ + kh * HD_;
    const float x1 = (float)k[d], x2 = (float)k[d + 64];
    _Float16* dst = ko + (((size_t)b * NKV_ + kh) * S_ + s) * HD_;
    dst[d]      = (_Float16)(x1 * cs - x2 * sn);
    dst[d + 64] = (_Float16)(x2 * cs + x1 * sn);
  }
}

// V head split: qkv16 -> v16 [B,NKV,S,HD]
__global__ __launch_bounds__(256)
void v_perm(const _Float16* __restrict__ qkv, _Float16* __restrict__ vo) {
  size_t i = (size_t)blockIdx.x * blockDim.x + threadIdx.x;
  const size_t total = (size_t)B_ * S_ * NKV_ * HD_;
  if (i >= total) return;
  const int d = (int)(i & 127);
  size_t t = i >> 7;
  const int kh = (int)(t % NKV_); t /= NKV_;
  const int s = (int)(t % S_);
  const int b = (int)(t / S_);
  vo[(((size_t)b * NKV_ + kh) * S_ + s) * HD_ + d] =
      qkv[((size_t)b * S_ + s) * QKVN_ + QS_ + KVS_ + kh * HD_ + d];
}

// ---------------------------------------------------------------------------
// Causal flash attention, GQA (4 q-heads per kv-head).
// Block = 128 threads (4 waves); q-tile 64 rows (16/wave); 32-key KV blocks.
// The KV loop is split: full (unmasked) blocks run without any causal
// compare/select VALU work; only the two diagonal blocks pay for masking.
// ---------------------------------------------------------------------------
__global__ __launch_bounds__(128)
void flash_attn(const _Float16* __restrict__ q, const _Float16* __restrict__ k,
                const _Float16* __restrict__ v, _Float16* __restrict__ o) {
  __shared__ __align__(32) _Float16 Vs[HD_ * 32];     // [hd][kswz(key)]
  __shared__ __align__(32) _Float16 Ps[4][16 * 32];   // per-wave P tile

  const int tid  = threadIdx.x;
  const int wave = tid >> 5;
  const int lane = tid & 31;
  const int half = lane >> 4;
  const int l16  = lane & 15;

  const int qt = blockIdx.x, h = blockIdx.y, b = blockIdx.z;
  const int kvh = h >> 2;
  const int qrow0 = qt * 64 + wave * 16;

  const _Float16* Qp = q + (((size_t)b * NH_ + h) * S_ + qrow0) * HD_;
  const _Float16* Kp = k + ((size_t)b * NKV_ + kvh) * S_ * HD_;
  const _Float16* Vp = v + ((size_t)b * NKV_ + kvh) * S_ * HD_;

  // Preload Q fragments for all 4 HD chunks (A-matrix layout).
  v16h qf[4];
#pragma unroll
  for (int c = 0; c < 4; ++c) {
    const v8h lo = *(const v8h*)(Qp + (size_t)l16 * HD_ + c * 32 + half * 8);
    const v8h hi = *(const v8h*)(Qp + (size_t)l16 * HD_ + c * 32 + 16 + half * 8);
#pragma unroll
    for (int e = 0; e < 8; ++e) { qf[c][e] = lo[e]; qf[c][8 + e] = hi[e]; }
  }

  v8f oacc[8] = {};
  float mrow[8], lrow[8];
#pragma unroll
  for (int j = 0; j < 8; ++j) { mrow[j] = -1e30f; lrow[j] = 0.f; }

  const float scale = 0.088388347648318447f;  // 1/sqrt(128)
  const int pw0 = kswz(l16);   // P write slot for key l16 (pair slot = +8)

  auto kvblock = [&](int kb, auto masked_t) {
    constexpr bool MASKED = decltype(masked_t)::value;

    // Stage V block (32 keys x 128 hd) transposed + k-swizzled into LDS.
    {
      const int kr = tid & 31, cb = (tid >> 5) * 32;
      const int kp = kswz(kr);
#pragma unroll
      for (int cc = 0; cc < 32; cc += 8) {
        v8h vv = *(const v8h*)(Vp + (size_t)(kb + kr) * HD_ + cb + cc);
#pragma unroll
        for (int e = 0; e < 8; ++e) Vs[(cb + cc + e) * 32 + kp] = vv[e];
      }
    }
    __syncthreads();

    // Scores: S[16q x 32k], two 16x16 C tiles, 8 WMMAs.
    v8f s0 = {}, s1 = {};
#pragma unroll
    for (int c = 0; c < 4; ++c) {
      v16h b0, b1;
      const _Float16* kp0 = Kp + (size_t)(kb + l16) * HD_ + c * 32 + half * 8;
      const _Float16* kp1 = Kp + (size_t)(kb + 16 + l16) * HD_ + c * 32 + half * 8;
      const v8h a0 = *(const v8h*)kp0, a1 = *(const v8h*)(kp0 + 16);
      const v8h c0 = *(const v8h*)kp1, c1 = *(const v8h*)(kp1 + 16);
#pragma unroll
      for (int e = 0; e < 8; ++e) {
        b0[e] = a0[e]; b0[8 + e] = a1[e];
        b1[e] = c0[e]; b1[8 + e] = c1[e];
      }
      s0 = __builtin_amdgcn_wmma_f32_16x16x32_f16(false, qf[c], false, b0, (short)0, s0, false, false);
      s1 = __builtin_amdgcn_wmma_f32_16x16x32_f16(false, qf[c], false, b1, (short)0, s1, false, false);
    }

    // Online softmax per row (row spans 16 lanes of one half).
#pragma unroll
    for (int j = 0; j < 8; ++j) {
      float x0, x1, p0, p1;
      if constexpr (MASKED) {
        const int qr = qrow0 + j + half * 8;
        const int key0 = kb + l16, key1 = kb + 16 + l16;
        x0 = (key0 <= qr) ? s0[j] * scale : -3.0e38f;
        x1 = (key1 <= qr) ? s1[j] * scale : -3.0e38f;
        float rm = fmaxf(x0, x1);
#pragma unroll
        for (int dd = 1; dd < 16; dd <<= 1) rm = fmaxf(rm, __shfl_xor(rm, dd, 16));
        const float mn = fmaxf(mrow[j], rm);
        const float alpha = __expf(mrow[j] - mn);
        p0 = (key0 <= qr) ? __expf(x0 - mn) : 0.f;
        p1 = (key1 <= qr) ? __expf(x1 - mn) : 0.f;
        float rs = p0 + p1;
#pragma unroll
        for (int dd = 1; dd < 16; dd <<= 1) rs += __shfl_xor(rs, dd, 16);
        lrow[j] = lrow[j] * alpha + rs;
        mrow[j] = mn;
#pragma unroll
        for (int t = 0; t < 8; ++t) oacc[t][j] *= alpha;
      } else {
        x0 = s0[j] * scale;
        x1 = s1[j] * scale;
        float rm = fmaxf(x0, x1);
#pragma unroll
        for (int dd = 1; dd < 16; dd <<= 1) rm = fmaxf(rm, __shfl_xor(rm, dd, 16));
        const float mn = fmaxf(mrow[j], rm);
        const float alpha = __expf(mrow[j] - mn);
        p0 = __expf(x0 - mn);
        p1 = __expf(x1 - mn);
        float rs = p0 + p1;
#pragma unroll
        for (int dd = 1; dd < 16; dd <<= 1) rs += __shfl_xor(rs, dd, 16);
        lrow[j] = lrow[j] * alpha + rs;
        mrow[j] = mn;
#pragma unroll
        for (int t = 0; t < 8; ++t) oacc[t][j] *= alpha;
      }
      Ps[wave][(j + half * 8) * 32 + pw0]     = (_Float16)p0;
      Ps[wave][(j + half * 8) * 32 + pw0 + 8] = (_Float16)p1;
    }
    wait_ds0();  // same-wave cross-lane LDS RAW before fragment re-read

    // P as A-matrix fragment: one contiguous 32B read per lane.
    const v16h pf = *(const v16h*)(&Ps[wave][l16 * 32 + half * 16]);

    // O += P (16x32) x V (32x128): 8 WMMAs.
#pragma unroll
    for (int t = 0; t < 8; ++t) {
      const v16h bf = *(const v16h*)(&Vs[(t * 16 + l16) * 32 + half * 16]);
      oacc[t] = __builtin_amdgcn_wmma_f32_16x16x32_f16(
          false, pf, false, bf, (short)0, oacc[t], false, false);
    }
    __syncthreads();
  };

  const int kdiag = qt * 64;
  int kb = 0;
  for (; kb < kdiag; kb += 32) kvblock(kb, std::false_type{});   // full blocks
  for (; kb < kdiag + 64; kb += 32) kvblock(kb, std::true_type{});  // diagonal

  // Normalize and write to [B, S, NH*HD] f16.
  _Float16* Op = o + (size_t)b * S_ * (NH_ * HD_) + (size_t)h * HD_;
#pragma unroll
  for (int t = 0; t < 8; ++t)
#pragma unroll
    for (int j = 0; j < 8; ++j) {
      const int qr = qrow0 + j + half * 8;
      Op[(size_t)qr * (NH_ * HD_) + t * 16 + l16] =
          (_Float16)(oacc[t][j] / lrow[j]);
    }
}

// ---------------------------------------------------------------------------
extern "C" void kernel_launch(void* const* d_in, const int* in_sizes, int n_in,
                              void* d_out, int out_size, void* d_ws, size_t ws_size,
                              hipStream_t stream) {
  const int*   positions = (const int*)d_in[0];
  const float* hidden    = (const float*)d_in[1];
  const float* w_qkv     = (const float*)d_in[2];
  const float* w_o       = (const float*)d_in[3];
  float* out = (float*)d_out;

  char* ws = (char*)d_ws;
  size_t off = 0;
  auto take = [&](size_t bytes) -> char* {
    char* p = ws + off;
    off += (bytes + 255) & ~(size_t)255;
    return p;
  };

  _Float16* h16   = (_Float16*)take((size_t)MTOT * H_ * 2);            // 32 MB (reused as attn out)
  _Float16* wq16  = (_Float16*)take((size_t)H_ * QKVN_ * 2);           // 48 MB (reused as w_o f16)
  _Float16* qkv16 = (_Float16*)take((size_t)MTOT * QKVN_ * 2);         // 48 MB
  _Float16* q16   = (_Float16*)take((size_t)B_ * NH_ * S_ * HD_ * 2);  // 32 MB
  _Float16* k16   = (_Float16*)take((size_t)B_ * NKV_ * S_ * HD_ * 2); //  8 MB
  _Float16* v16p  = (_Float16*)take((size_t)B_ * NKV_ * S_ * HD_ * 2); //  8 MB
  _Float16* attn16 = h16;   // safe: h16 consumed by QKV GEMM before attention
  _Float16* wo16   = wq16;  // safe: wq16 consumed by QKV GEMM before conversion

  {
    size_t n = (size_t)MTOT * H_;
    cvt_f32_f16<<<dim3((unsigned)((n / 4 + 255) / 256)), 256, 0, stream>>>(hidden, h16, n);
  }
  {
    size_t n = (size_t)H_ * QKVN_;
    cvt_f32_f16<<<dim3((unsigned)((n / 4 + 255) / 256)), 256, 0, stream>>>(w_qkv, wq16, n);
  }

  gemm_wmma<1><<<dim3(QKVN_ / 128, MTOT / 128), 256, 0, stream>>>(
      h16, wq16, (void*)qkv16, MTOT, QKVN_, H_);

  {
    size_t n = (size_t)H_ * H_;
    cvt_f32_f16<<<dim3((unsigned)((n / 4 + 255) / 256)), 256, 0, stream>>>(w_o, wo16, n);
  }

  {
    size_t n = (size_t)B_ * S_ * (NH_ + NKV_) * 64;
    rope_split<<<dim3((unsigned)((n + 255) / 256)), 256, 0, stream>>>(
        positions, qkv16, q16, k16);
  }
  {
    size_t n = (size_t)B_ * S_ * NKV_ * HD_;
    v_perm<<<dim3((unsigned)((n + 255) / 256)), 256, 0, stream>>>(qkv16, v16p);
  }

  flash_attn<<<dim3(S_ / 64, NH_, B_), 128, 0, stream>>>(q16, k16, v16p, attn16);

  gemm_wmma<0><<<dim3(H_ / 128, MTOT / 128), 256, 0, stream>>>(
      attn16, wo16, (void*)out, MTOT, H_, H_);
}
